// RNN_31945966747760
// MI455X (gfx1250) — matrix operand order repaired and checked by reading, not compiled, not run on hardware
//
#include <hip/hip_runtime.h>
#include <hip/hip_bf16.h>

#define T_STEPS 256
#define BATCH   128
#define HID     1024
#define LAYERS  4
#define NTILES  (HID / 16)    // 64 column tiles
#define MTILES  (BATCH / 16)  // 8 row tiles

typedef __attribute__((ext_vector_type(16))) __bf16 v16bf;
typedef __attribute__((ext_vector_type(8)))  __bf16 v8bf;
typedef __attribute__((ext_vector_type(8)))  float  v8f;

// Branch-free tanh: t = exp2(-2*log2(e)*|x|) in (0,1]; tanh = sign(x)*(1-t)/(1+t).
// One v_exp_f32 + one v_rcp_f32, no divergence, saturates cleanly (t->0 => 1).
__device__ __forceinline__ float fast_tanh(float x) {
    float ax = __builtin_fabsf(x);
    float t  = __builtin_amdgcn_exp2f(ax * -2.88539008177792681472f);
    float r  = (1.0f - t) * __builtin_amdgcn_rcpf(1.0f + t);
    return __builtin_copysignf(r, x);
}

// A-matrix (16x32 bf16) fragment load, contiguous k order:
// lane l (m = l%16, half = l/16) holds A[m][16*half + 0..7] in elems [0..7]
// and A[m][16*half + 8 + 0..7] in elems [8..15].
__device__ __forceinline__ v16bf load_a_frag(const __bf16* __restrict__ row,
                                             int k0, int half) {
    v8bf lo = *(const v8bf*)(row + k0 + 16 * half);
    v8bf hi = *(const v8bf*)(row + k0 + 16 * half + 8);
    v16bf a;
#pragma unroll
    for (int i = 0; i < 8; ++i) { a[i] = lo[i]; a[i + 8] = hi[i]; }
    return a;
}

// fp32 -> bf16 bulk convert (grid-stride).
__global__ void f32_to_bf16_kernel(const float* __restrict__ src,
                                   __bf16* __restrict__ dst, size_t n) {
    size_t i = (size_t)blockIdx.x * blockDim.x + threadIdx.x;
    size_t stride = (size_t)gridDim.x * blockDim.x;
    for (; i < n; i += stride) dst[i] = (__bf16)src[i];
}

// bias = b_ih + b_hh (combined once); also zero the grid-barrier counters.
__global__ void prep_kernel(const float* __restrict__ b_ih,
                            const float* __restrict__ b_hh,
                            float* __restrict__ bias,
                            unsigned* __restrict__ barriers, int nbias) {
    int i = blockIdx.x * blockDim.x + threadIdx.x;
    if (i < nbias) bias[i] = b_ih[i] + b_hh[i];
    if (i < 256) barriers[i] = 0u;
}

// Persistent per-layer RNN kernel.
// grid = 64 WGs (one per 16-wide output column tile), block = 256 (8 waves,
// one per 16-row batch tile). Weight slices live in LDS for all 256 steps.
// Steps are separated by an agent-scope atomic grid barrier.
__global__ void __launch_bounds__(256)
rnn_layer_kernel(const __bf16* __restrict__ actIn,   // [T,B,H] bf16
                 __bf16* __restrict__ actOut,        // [T,B,H] bf16
                 const __bf16* __restrict__ Wi,      // [H,H] bf16 row-major
                 const __bf16* __restrict__ Wh,      // [H,H] bf16 row-major
                 const __bf16* __restrict__ h0,      // [B,H] bf16
                 const float* __restrict__ bias,     // [H] fp32 (b_ih+b_hh)
                 float* __restrict__ outF32,         // [T,B,H] fp32 or null
                 float* __restrict__ hnsOut,         // [B,H] fp32 (h at t=T-1)
                 unsigned* __restrict__ barrier) {
    __shared__ __bf16 ldsWi[16 * HID];  // 32 KB: Wi rows [tileN*16, +16)
    __shared__ __bf16 ldsWh[16 * HID];  // 32 KB: Wh rows [tileN*16, +16)

    const int tid   = threadIdx.x;
    const int wave  = tid >> 5;
    const int lane  = tid & 31;
    const int tileN = blockIdx.x;  // 0..63 output column tile
    const int tileM = wave;        // 0..7  batch row tile

    // Stage this WG's 16-row weight slices into LDS (once; reused 256 steps).
    {
        const uint4* sWi = (const uint4*)(Wi + (size_t)tileN * 16 * HID);
        const uint4* sWh = (const uint4*)(Wh + (size_t)tileN * 16 * HID);
        uint4* dWi = (uint4*)ldsWi;
        uint4* dWh = (uint4*)ldsWh;
        for (int i = tid; i < 16 * HID / 8; i += 256) {
            dWi[i] = sWi[i];
            dWh[i] = sWh[i];
        }
    }
    __syncthreads();

    const int nn   = lane & 15;            // A row within tile, B col, D col
    const int half = lane >> 4;
    const int rowb = tileM * 16 + nn;      // batch row this lane loads (A)
    const int col  = tileN * 16 + nn;      // output column this lane owns (D)
    const float bv = bias[col];

    for (int t = 0; t < T_STEPS; ++t) {
        const __bf16* xrow = actIn + ((size_t)t * BATCH + rowb) * HID;
        const __bf16* hrow =
            (t == 0) ? (h0 + (size_t)rowb * HID)
                     : (actOut + ((size_t)(t - 1) * BATCH + rowb) * HID);

        v8f acc = {};
#pragma unroll 4
        for (int k0 = 0; k0 < HID; k0 += 32) {
            // x_t @ Wi^T contribution
            v16bf ax = load_a_frag(xrow, k0, half);
            v16bf bi = *(const v16bf*)(ldsWi + nn * HID + k0 + 16 * half);
            acc = __builtin_amdgcn_wmma_f32_16x16x32_bf16(
                false, ax, false, bi, (short)0, acc, false, false);
            // h_{t-1} @ Wh^T contribution
            v16bf ah = load_a_frag(hrow, k0, half);
            v16bf bh = *(const v16bf*)(ldsWh + nn * HID + k0 + 16 * half);
            acc = __builtin_amdgcn_wmma_f32_16x16x32_bf16(
                false, ah, false, bh, (short)0, acc, false, false);
        }

        // D element i of this lane is (M = i + 8*half, N = nn) of the tile.
#pragma unroll
        for (int i = 0; i < 8; ++i) {
            const int orow = tileM * 16 + 8 * half + i;
            const float v  = fast_tanh(acc[i] + bv);
            const size_t oidx = ((size_t)t * BATCH + orow) * HID + col;
            actOut[oidx] = (__bf16)v;
            if (outF32) outF32[oidx] = v;
            if (t == T_STEPS - 1) hnsOut[(size_t)orow * HID + col] = v;
        }

        // Grid-wide barrier: monotonically increasing agent-scope counter.
        __syncthreads();
        if (tid == 0) {
            __hip_atomic_fetch_add(barrier, 1u, __ATOMIC_RELEASE,
                                   __HIP_MEMORY_SCOPE_AGENT);
            const unsigned target = (unsigned)gridDim.x * (unsigned)(t + 1);
            while (__hip_atomic_load(barrier, __ATOMIC_ACQUIRE,
                                     __HIP_MEMORY_SCOPE_AGENT) < target)
                __builtin_amdgcn_s_sleep(8);
        }
        __syncthreads();
        __threadfence();  // make other WGs' h_t stores visible to all lanes
    }
}

extern "C" void kernel_launch(void* const* d_in, const int* in_sizes, int n_in,
                              void* d_out, int out_size, void* d_ws,
                              size_t ws_size, hipStream_t stream) {
    const float* inputs = (const float*)d_in[0];  // [T,B,H]
    const float* hxs    = (const float*)d_in[1];  // [L,B,H]
    const float* W_ih   = (const float*)d_in[2];  // [L,H,H]
    const float* b_ih   = (const float*)d_in[3];  // [L,H]
    const float* W_hh   = (const float*)d_in[4];  // [L,H,H]
    const float* b_hh   = (const float*)d_in[5];  // [L,H]

    float* outX = (float*)d_out;                           // [T,B,H]
    float* outH = outX + (size_t)T_STEPS * BATCH * HID;    // [L,B,H]

    const size_t ACT = (size_t)T_STEPS * BATCH * HID;  // 33,554,432
    char* ws = (char*)d_ws;
    __bf16* act0 = (__bf16*)ws;  ws += ACT * 2;                       // 64 MB
    __bf16* act1 = (__bf16*)ws;  ws += ACT * 2;                       // 64 MB
    __bf16* Wib  = (__bf16*)ws;  ws += (size_t)LAYERS * HID * HID * 2;  // 8 MB
    __bf16* Whb  = (__bf16*)ws;  ws += (size_t)LAYERS * HID * HID * 2;  // 8 MB
    __bf16* hxsb = (__bf16*)ws;  ws += (size_t)LAYERS * BATCH * HID * 2; // 1 MB
    float*  bias = (float*)ws;   ws += (size_t)LAYERS * HID * 4;       // 16 KB
    unsigned* barriers = (unsigned*)ws;                                 // 1 KB

    auto cvt = [&](const float* s, __bf16* d, size_t n) {
        int blocks = (int)((n + 256 * 8 - 1) / (256 * 8));
        f32_to_bf16_kernel<<<blocks, 256, 0, stream>>>(s, d, n);
    };
    cvt(inputs, act0, ACT);
    cvt(hxs, hxsb, (size_t)LAYERS * BATCH * HID);
    cvt(W_ih, Wib, (size_t)LAYERS * HID * HID);
    cvt(W_hh, Whb, (size_t)LAYERS * HID * HID);
    prep_kernel<<<(LAYERS * HID + 255) / 256, 256, 0, stream>>>(
        b_ih, b_hh, bias, barriers, LAYERS * HID);

    for (int l = 0; l < LAYERS; ++l) {
        const __bf16* in = (l & 1) ? act1 : act0;
        __bf16* out      = (l & 1) ? act0 : act1;
        rnn_layer_kernel<<<NTILES, 256, 0, stream>>>(
            in, out, Wib + (size_t)l * HID * HID, Whb + (size_t)l * HID * HID,
            hxsb + (size_t)l * BATCH * HID, bias + (size_t)l * HID,
            (l == LAYERS - 1) ? outX : nullptr,
            outH + (size_t)l * BATCH * HID, barriers + l * 64);
    }
}
